// Metalayer_52733608460305
// MI455X (gfx1250) — matrix operand order repaired and checked by reading, not compiled, not run on hardware
//
#include <hip/hip_runtime.h>
#include <hip/hip_bf16.h>
#include <math.h>

typedef __attribute__((ext_vector_type(16))) _Float16 v16h;
typedef __attribute__((ext_vector_type(8)))  _Float16 half8;
typedef __attribute__((ext_vector_type(2)))  _Float16 h2pk;
typedef __attribute__((ext_vector_type(8)))  float    v8f;

#define NGRID   96
#define NSQ     9216
#define NOFFS   13
#define NNZ     119808
#define NODES   128
#define OUTRES  8
#define SIDE    816
#define BORDER  24
#define KCONST  6.68423968753147f   /* 2*pi/0.94 */

// ---------------- workspace layout (floats) ----------------
#define WS_HS      0
#define WS_NEFFS   9216
#define WS_EYS     18432
#define WS_U0      27648
#define WS_URE     36864
#define WS_UIM     46080
#define WS_MVALS   55296
#define WS_CINV    175104
#define WS_W2F16   294912   /* 3 * 16384 halves = 24576 floats */

// CDNA5 hardware tanh (VALU trans op).  Prefer the builtin (compiler manages
// the trans-op RAW hazard); fall back to inline asm with an embedded v_nop.
__device__ __forceinline__ float fast_tanh(float x)
{
#if __has_builtin(__builtin_amdgcn_tanhf)
    return __builtin_amdgcn_tanhf(x);
#elif __has_builtin(__builtin_amdgcn_tanh_f32)
    return __builtin_amdgcn_tanh_f32(x);
#else
    float y;
    asm("v_tanh_f32 %0, %1\n\tv_nop" : "=v"(y) : "v"(x));
    return y;
#endif
}

// =====================================================================
// Kernel 1: scalars  hs, Eys, U0 = pooled(E0) * (0.5+0.5*sigmoid(hs))
// =====================================================================
__global__ void prep_scalars_kernel(const float* __restrict__ h_paras,
                                    const float* __restrict__ E0,
                                    float* __restrict__ hs,
                                    float* __restrict__ Eys,
                                    float* __restrict__ U0)
{
    int tid = blockIdx.x * blockDim.x + threadIdx.x;
    if (tid >= NSQ) return;
    float hp  = h_paras[tid];
    float sg  = 1.0f / (1.0f + __expf(-hp));
    float h   = sg * 0.6f + 0.2f;
    hs[tid]   = h;
    Eys[tid]  = 1.0f + 0.1f * fast_tanh(h);
    float coupl = 0.5f + 0.5f / (1.0f + __expf(-h));
    int i = tid / NGRID, j = tid % NGRID;
    const float* base = E0 + (size_t)(BORDER + i * OUTRES) * SIDE + (BORDER + j * OUTRES);
    float s = 0.0f;
    #pragma unroll
    for (int p = 0; p < OUTRES; ++p) {
        #pragma unroll
        for (int q = 0; q < OUTRES; ++q) s += base[p * SIDE + q];
    }
    U0[tid] = (s * (1.0f / 64.0f)) * coupl;
}

// =====================================================================
// Kernel 2: repack w2n/w2c/w2k (f32, 128x128 row-major [K][N]) into f16
// WMMA B-fragment order: dst[((kt*8+nt)*32+lane)*16 + e] =
//     W[32*kt + e + 16*(lane/16)][16*nt + lane%16]
// =====================================================================
__global__ void prep_weights_kernel(const float* __restrict__ w2n,
                                    const float* __restrict__ w2c,
                                    const float* __restrict__ w2k,
                                    _Float16* __restrict__ w2n16,
                                    _Float16* __restrict__ w2c16,
                                    _Float16* __restrict__ w2k16)
{
    int tid = blockIdx.x * blockDim.x + threadIdx.x;
    if (tid >= 3 * 16384) return;
    int mat  = tid >> 14;
    int rem  = tid & 16383;
    int e    = rem & 15;
    int lane = (rem >> 4) & 31;
    int frag = rem >> 9;            // 0..31
    int nt   = frag & 7;
    int kt   = frag >> 3;
    int K    = kt * 32 + e + ((lane >> 4) << 4);
    int Ncol = nt * 16 + (lane & 15);
    const float* W = (mat == 0) ? w2n : (mat == 1) ? w2c : w2k;
    _Float16*    D = (mat == 0) ? w2n16 : (mat == 1) ? w2c16 : w2k16;
    D[rem] = (_Float16)W[K * NODES + Ncol];
}

// =====================================================================
// WMMA helpers
// =====================================================================
__device__ __forceinline__ v16h load_a_frag(const _Float16* h1w, int mrow, int g, int kt)
{
    // A(16x32 f16): lane L holds row M=L%16; element e -> K=16g+e
    const _Float16* p = h1w + mrow * NODES + kt * 32 + g * 16;
    half8 lo = *reinterpret_cast<const half8*>(p);
    half8 hi = *reinterpret_cast<const half8*>(p + 8);
    return __builtin_shufflevector(lo, hi, 0,1,2,3,4,5,6,7,8,9,10,11,12,13,14,15);
}

// layer2 (WMMA, 128x128) + tanh + layer3 (dot w3) for a 32-row tile.
// Two A-row-sets (rows 0-15 and 16-31) share every B fragment load.
// The nt loop is kept ROLLED (unroll 1) so only one accumulator pair is live:
// full unrolling makes the scheduler keep 16 v8f accumulators -> scratch spills.
// b2/w3 scalars are loaded per-iteration from global (cached) because a rolled
// loop would otherwise dynamically index a register array.
// After the 16-lane xor-reduce every lane holds:
//   res0[r] = row ( 8*(lane/16)+r)  pre-bias output
//   res1[r] = row (16+8*(lane/16)+r) pre-bias output
__device__ __forceinline__ void mlp32_l2_l3(const _Float16* __restrict__ h1w,
                                            const _Float16* __restrict__ w2lds,
                                            const float* __restrict__ b2g,
                                            const float* __restrict__ w3g,
                                            int lane, float res0[8], float res1[8])
{
    const int g = lane >> 4, mrow = lane & 15;
    v16h a0[4], a1[4];
    #pragma unroll
    for (int kt = 0; kt < 4; ++kt) {
        a0[kt] = load_a_frag(h1w,              mrow, g, kt);
        a1[kt] = load_a_frag(h1w + 16 * NODES, mrow, g, kt);
    }
    float acc0[8] = {0,0,0,0,0,0,0,0};
    float acc1[8] = {0,0,0,0,0,0,0,0};
    #pragma unroll 1
    for (int nt = 0; nt < 8; ++nt) {
        v8f c0 = {};
        v8f c1 = {};
        #pragma unroll
        for (int kt = 0; kt < 4; ++kt) {
            v16h b = *reinterpret_cast<const v16h*>(w2lds + (kt * 8 + nt) * 512 + lane * 16);
            c0 = __builtin_amdgcn_wmma_f32_16x16x32_f16(false, a0[kt], false, b,
                                                        (short)0, c0, false, false);
            c1 = __builtin_amdgcn_wmma_f32_16x16x32_f16(false, a1[kt], false, b,
                                                        (short)0, c1, false, false);
        }
        float b2v = b2g[nt * 16 + mrow];
        float w3v = w3g[nt * 16 + mrow];
        #pragma unroll
        for (int r = 0; r < 8; ++r) {
            acc0[r] = fmaf(fast_tanh(c0[r] + b2v), w3v, acc0[r]);
            acc1[r] = fmaf(fast_tanh(c1[r] + b2v), w3v, acc1[r]);
        }
    }
    #pragma unroll
    for (int m = 1; m < 16; m <<= 1) {
        #pragma unroll
        for (int r = 0; r < 8; ++r) {
            acc0[r] += __shfl_xor(acc0[r], m, 32);
            acc1[r] += __shfl_xor(acc1[r], m, 32);
        }
    }
    #pragma unroll
    for (int r = 0; r < 8; ++r) { res0[r] = acc0[r]; res1[r] = acc1[r]; }
}

// =====================================================================
// Kernel 3: node MLP  (9216 rows, 1 -> 128 -> 128 -> 1) => neffs
// 32 rows per wave-tile
// =====================================================================
__global__ void node_mlp_kernel(const float* __restrict__ hs,
                                const float* __restrict__ w1n, const float* __restrict__ b1n,
                                const float* __restrict__ b2n, const float* __restrict__ w3n,
                                const float* __restrict__ b3n,
                                const _Float16* __restrict__ w2n16,
                                float* __restrict__ neffs)
{
    extern __shared__ char smem[];
    _Float16* w2lds = reinterpret_cast<_Float16*>(smem);         // 16384 halves
    _Float16* h1all = w2lds + 16384;                             // 4 * 4096 halves
    const int lane = threadIdx.x & 31;
    const int wib  = threadIdx.x >> 5;
    _Float16* h1w  = h1all + wib * 4096;

    {   // stage W2 (32 KB) into LDS
        const uint4* src = reinterpret_cast<const uint4*>(w2n16);
        uint4* dst = reinterpret_cast<uint4*>(w2lds);
        for (int i = threadIdx.x; i < 2048; i += blockDim.x) dst[i] = src[i];
    }
    __syncthreads();

    const int g = lane >> 4, mrow = lane & 15;
    // layer1: lane owns column pairs (2*lane, 2*lane+1) and (+64)
    float w1r[4], b1r[4];
    #pragma unroll
    for (int t2 = 0; t2 < 4; ++t2) {
        int col = 2 * lane + 64 * (t2 >> 1) + (t2 & 1);
        w1r[t2] = w1n[col]; b1r[t2] = b1n[col];
    }
    const float b3v = b3n[0];

    const int gw = blockIdx.x * (blockDim.x >> 5) + wib;
    const int nw = gridDim.x * (blockDim.x >> 5);
    for (int tile = gw; tile < NSQ / 32; tile += nw) {
        const int i0 = tile * 32;
        float xv = hs[i0 + lane];                 // lane m owns node i0+m
        #pragma unroll
        for (int m = 0; m < 32; ++m) {
            float x = __shfl(xv, m, 32);
            #pragma unroll
            for (int t = 0; t < 2; ++t) {
                h2pk pk;
                pk[0] = (_Float16)fast_tanh(fmaf(x, w1r[2 * t + 0], b1r[2 * t + 0]));
                pk[1] = (_Float16)fast_tanh(fmaf(x, w1r[2 * t + 1], b1r[2 * t + 1]));
                *reinterpret_cast<h2pk*>(h1w + m * NODES + 2 * lane + 64 * t) = pk;
            }
        }
        asm volatile("s_wait_dscnt 0x0" ::: "memory");
        float res0[8], res1[8];
        mlp32_l2_l3(h1w, w2lds, b2n, w3n, lane, res0, res1);
        if (mrow == 0) {
            #pragma unroll
            for (int r = 0; r < 8; ++r) {
                neffs[i0 +      8 * g + r] = res0[r] + b3v;
                neffs[i0 + 16 + 8 * g + r] = res1[r] + b3v;
            }
        }
        asm volatile("s_wait_dscnt 0x0" ::: "memory");
    }
}

// =====================================================================
// Kernel 4: edge MLPs (119808 rows, 4->128->128->1, c and k) => m_vals, cinv
// 32 edges per wave-tile
// =====================================================================
__global__ void edge_mlp_kernel(const float* __restrict__ hs,
                                const float* __restrict__ neffs,
                                const int*   __restrict__ coo,
                                const float* __restrict__ dis,
                                const float* __restrict__ mask,
                                const float* __restrict__ is_center,
                                const float* __restrict__ w1c, const float* __restrict__ b1c,
                                const float* __restrict__ b2c, const float* __restrict__ w3c,
                                const float* __restrict__ b3c,
                                const float* __restrict__ w1k, const float* __restrict__ b1k,
                                const float* __restrict__ b2k, const float* __restrict__ w3k,
                                const float* __restrict__ b3k,
                                const _Float16* __restrict__ w2c16,   // c then k contiguous
                                float* __restrict__ m_vals,
                                float* __restrict__ cinv_vals)
{
    extern __shared__ char smem[];
    _Float16* w2c_lds = reinterpret_cast<_Float16*>(smem);       // 16384 halves
    _Float16* w2k_lds = w2c_lds + 16384;                         // 16384 halves
    _Float16* h1all   = w2k_lds + 16384;                         // 4 * 4096 halves
    const int lane = threadIdx.x & 31;
    const int wib  = threadIdx.x >> 5;
    _Float16* h1w  = h1all + wib * 4096;

    {   // stage W2c + W2k (64 KB) into LDS
        const uint4* src = reinterpret_cast<const uint4*>(w2c16);
        uint4* dst = reinterpret_cast<uint4*>(w2c_lds);
        for (int i = threadIdx.x; i < 4096; i += blockDim.x) dst[i] = src[i];
    }
    __syncthreads();

    const int g = lane >> 4, mrow = lane & 15;
    // layer1: lane owns column pairs (2*lane, 2*lane+1) and (+64)
    float w1cr[16], b1cr[4], w1kr[16], b1kr[4];
    #pragma unroll
    for (int t2 = 0; t2 < 4; ++t2) {
        int col = 2 * lane + 64 * (t2 >> 1) + (t2 & 1);
        #pragma unroll
        for (int f = 0; f < 4; ++f) { w1cr[f * 4 + t2] = w1c[f * NODES + col];
                                      w1kr[f * 4 + t2] = w1k[f * NODES + col]; }
        b1cr[t2] = b1c[col]; b1kr[t2] = b1k[col];
    }
    const float b3cv = b3c[0], b3kv = b3k[0];

    const int gw = blockIdx.x * (blockDim.x >> 5) + wib;
    const int nw = gridDim.x * (blockDim.x >> 5);
    for (int tile = gw; tile < NNZ / 32; tile += nw) {
        const int e0 = tile * 32;

        // lane m owns edge e0+m's 4 inputs; rows broadcast via shfl.
        // Registers reused by both the c and k pipelines.
        const int el = e0 + lane;
        float x0v = hs[coo[el]];
        float x1v = hs[coo[NNZ + el]];
        float x2v = dis[2 * el];
        float x3v = dis[2 * el + 1];

        // ---------- c pipeline ----------
        #pragma unroll
        for (int m = 0; m < 32; ++m) {
            float x0 = __shfl(x0v, m, 32);
            float x1 = __shfl(x1v, m, 32);
            float x2 = __shfl(x2v, m, 32);
            float x3 = __shfl(x3v, m, 32);
            #pragma unroll
            for (int t = 0; t < 2; ++t) {
                h2pk pk;
                #pragma unroll
                for (int hlf = 0; hlf < 2; ++hlf) {
                    int t2 = 2 * t + hlf;
                    float h = b1cr[t2];
                    h = fmaf(x0, w1cr[0 * 4 + t2], h);
                    h = fmaf(x1, w1cr[1 * 4 + t2], h);
                    h = fmaf(x2, w1cr[2 * 4 + t2], h);
                    h = fmaf(x3, w1cr[3 * 4 + t2], h);
                    pk[hlf] = (_Float16)fast_tanh(h);
                }
                *reinterpret_cast<h2pk*>(h1w + m * NODES + 2 * lane + 64 * t) = pk;
            }
        }
        asm volatile("s_wait_dscnt 0x0" ::: "memory");
        float cres0[8], cres1[8];
        mlp32_l2_l3(h1w, w2c_lds, b2c, w3c, lane, cres0, cres1);
        asm volatile("s_wait_dscnt 0x0" ::: "memory");

        // ---------- k pipeline (reuse h1 buffer + input regs) ----------
        #pragma unroll
        for (int m = 0; m < 32; ++m) {
            float x0 = __shfl(x0v, m, 32);
            float x1 = __shfl(x1v, m, 32);
            float x2 = __shfl(x2v, m, 32);
            float x3 = __shfl(x3v, m, 32);
            #pragma unroll
            for (int t = 0; t < 2; ++t) {
                h2pk pk;
                #pragma unroll
                for (int hlf = 0; hlf < 2; ++hlf) {
                    int t2 = 2 * t + hlf;
                    float h = b1kr[t2];
                    h = fmaf(x0, w1kr[0 * 4 + t2], h);
                    h = fmaf(x1, w1kr[1 * 4 + t2], h);
                    h = fmaf(x2, w1kr[2 * 4 + t2], h);
                    h = fmaf(x3, w1kr[3 * 4 + t2], h);
                    pk[hlf] = (_Float16)fast_tanh(h);
                }
                *reinterpret_cast<h2pk*>(h1w + m * NODES + 2 * lane + 64 * t) = pk;
            }
        }
        asm volatile("s_wait_dscnt 0x0" ::: "memory");
        float kres0[8], kres1[8];
        mlp32_l2_l3(h1w, w2k_lds, b2k, w3k, lane, kres0, kres1);
        asm volatile("s_wait_dscnt 0x0" ::: "memory");

        // ---------- postprocess (lanes 0 and 16 write 16 edges each) ----------
        if (mrow == 0) {
            #pragma unroll
            for (int s = 0; s < 2; ++s) {
                #pragma unroll
                for (int r = 0; r < 8; ++r) {
                    int m = 16 * s + 8 * g + r;
                    int e = e0 + m;
                    float craw = (s ? cres1[r] : cres0[r]) + b3cv;
                    float kraw = (s ? kres1[r] : kres0[r]) + b3kv;
                    float msk  = mask[e];
                    float ic   = is_center[e];
                    float cv   = msk * 0.05f * fast_tanh(craw) + ic;
                    float kv   = msk * 0.10f * fast_tanh(kraw);
                    float ne   = neffs[coo[e]];
                    m_vals[e]    = KCONST * ne * cv + kv;
                    cinv_vals[e] = 2.0f * ic - cv;
                }
            }
        }
    }
}

// =====================================================================
// Kernel 5: Euler loop, fully LDS-resident (1 workgroup, 1024 threads)
//   u <- u + i*dz * spmv(cinv, spmv(m, u)), coefficients real
// =====================================================================
__global__ void euler_kernel(const float* __restrict__ U0,
                             const float* __restrict__ mv,
                             const float* __restrict__ cv,
                             const int*   __restrict__ stepsPtr,
                             float* __restrict__ uReOut,
                             float* __restrict__ uImOut)
{
    extern __shared__ char smem[];
    float* uR = reinterpret_cast<float*>(smem);
    float* uI = uR + NSQ;
    float* tR = uI + NSQ;
    float* tI = tR + NSQ;

    const int OI[NOFFS] = {0, 1,-1, 0, 0, 1, 1,-1,-1, 2,-2, 0, 0};
    const int OJ[NOFFS] = {0, 0, 0, 1,-1, 1,-1, 1,-1, 0, 0, 2,-2};

    const int tid = threadIdx.x;
    #pragma unroll
    for (int q = 0; q < 9; ++q) {
        int c = tid + q * 1024;
        uR[c] = U0[c];
        uI[c] = 0.0f;
    }
    __syncthreads();

    const int steps = *stepsPtr;
    const float dz = 1.5f / (float)steps;

    for (int s = 0; s < steps; ++s) {
        #pragma unroll
        for (int q = 0; q < 9; ++q) {
            int c = tid + q * 1024;
            int i = c / NGRID, j = c % NGRID;
            float sr = 0.0f, si = 0.0f;
            #pragma unroll
            for (int t = 0; t < NOFFS; ++t) {
                int ni = i + OI[t]; ni = ni < 0 ? 0 : (ni > NGRID - 1 ? NGRID - 1 : ni);
                int nj = j + OJ[t]; nj = nj < 0 ? 0 : (nj > NGRID - 1 ? NGRID - 1 : nj);
                int nb = ni * NGRID + nj;
                float v = mv[c * NOFFS + t];
                sr = fmaf(v, uR[nb], sr);
                si = fmaf(v, uI[nb], si);
            }
            tR[c] = sr; tI[c] = si;
        }
        __syncthreads();
        #pragma unroll
        for (int q = 0; q < 9; ++q) {
            int c = tid + q * 1024;
            int i = c / NGRID, j = c % NGRID;
            float sr = 0.0f, si = 0.0f;
            #pragma unroll
            for (int t = 0; t < NOFFS; ++t) {
                int ni = i + OI[t]; ni = ni < 0 ? 0 : (ni > NGRID - 1 ? NGRID - 1 : ni);
                int nj = j + OJ[t]; nj = nj < 0 ? 0 : (nj > NGRID - 1 ? NGRID - 1 : nj);
                int nb = ni * NGRID + nj;
                float v = cv[c * NOFFS + t];
                sr = fmaf(v, tR[nb], sr);
                si = fmaf(v, tI[nb], si);
            }
            // u += i*dz*(sr + i*si) = (-dz*si) + i*(dz*sr)
            uR[c] = fmaf(-dz, si, uR[c]);
            uI[c] = fmaf( dz, sr, uI[c]);
        }
        __syncthreads();
    }

    #pragma unroll
    for (int q = 0; q < 9; ++q) {
        int c = tid + q * 1024;
        uReOut[c] = uR[c];
        uImOut[c] = uI[c];
    }
}

// =====================================================================
// Kernel 6: upsample  En[i*8+p][j*8+q] = (Uz*Eys)[i,j] * profile[p,q]
// out[0] = Re, out[1] = Im   (2 x 768 x 768)
// =====================================================================
__global__ void upsample_kernel(const float* __restrict__ uRe,
                                const float* __restrict__ uIm,
                                const float* __restrict__ Eys,
                                const float* __restrict__ prof_re,
                                const float* __restrict__ prof_im,
                                float* __restrict__ out)
{
    const int W = NGRID * OUTRES;              // 768
    int tid = blockIdx.x * blockDim.x + threadIdx.x;
    if (tid >= W * W) return;
    int x = tid % W, y = tid / W;
    int i = y >> 3, p = y & 7, j = x >> 3, q = x & 7;
    int c = i * NGRID + j;
    float ey = Eys[c];
    float gr = uRe[c] * ey;
    float gi = uIm[c] * ey;
    float pr = prof_re[p * OUTRES + q];
    float pi = prof_im[p * OUTRES + q];
    out[tid]         = gr * pr - gi * pi;      // real plane
    out[W * W + tid] = gr * pi + gi * pr;      // imag plane
}

// =====================================================================
// Launcher
// =====================================================================
extern "C" void kernel_launch(void* const* d_in, const int* in_sizes, int n_in,
                              void* d_out, int out_size, void* d_ws, size_t ws_size,
                              hipStream_t stream)
{
    const float* h_paras  = (const float*)d_in[0];
    const float* E0       = (const float*)d_in[1];
    const float* w1n      = (const float*)d_in[2];
    const float* b1n      = (const float*)d_in[3];
    const float* w2n      = (const float*)d_in[4];
    const float* b2n      = (const float*)d_in[5];
    const float* w3n      = (const float*)d_in[6];
    const float* b3n      = (const float*)d_in[7];
    const float* w1c      = (const float*)d_in[8];
    const float* b1c      = (const float*)d_in[9];
    const float* w2c      = (const float*)d_in[10];
    const float* b2c      = (const float*)d_in[11];
    const float* w3c      = (const float*)d_in[12];
    const float* b3c      = (const float*)d_in[13];
    const float* w1k      = (const float*)d_in[14];
    const float* b1k      = (const float*)d_in[15];
    const float* w2k      = (const float*)d_in[16];
    const float* b2k      = (const float*)d_in[17];
    const float* w3k      = (const float*)d_in[18];
    const float* b3k      = (const float*)d_in[19];
    const float* prof_re  = (const float*)d_in[20];
    const float* prof_im  = (const float*)d_in[21];
    const float* dis      = (const float*)d_in[22];
    const float* mask     = (const float*)d_in[23];
    const float* is_cent  = (const float*)d_in[24];
    const int*   coo      = (const int*)d_in[25];
    const int*   steps    = (const int*)d_in[26];

    float* ws = (float*)d_ws;
    float* hs     = ws + WS_HS;
    float* neffs  = ws + WS_NEFFS;
    float* Eys    = ws + WS_EYS;
    float* U0     = ws + WS_U0;
    float* uRe    = ws + WS_URE;
    float* uIm    = ws + WS_UIM;
    float* m_vals = ws + WS_MVALS;
    float* cinv   = ws + WS_CINV;
    _Float16* w2n16 = (_Float16*)(ws + WS_W2F16);
    _Float16* w2c16 = w2n16 + 16384;
    _Float16* w2k16 = w2n16 + 32768;

    // 1) scalars + pooling
    prep_scalars_kernel<<<NSQ / 256, 256, 0, stream>>>(h_paras, E0, hs, Eys, U0);
    // 2) repack W2 weights to f16 B-fragment order
    prep_weights_kernel<<<(3 * 16384) / 256, 256, 0, stream>>>(w2n, w2c, w2k,
                                                               w2n16, w2c16, w2k16);
    // 3) node MLP -> neffs   (LDS: 32KB weights + 4*8KB h1 tiles = 64KB)
    node_mlp_kernel<<<36, 128, 65536, stream>>>(hs, w1n, b1n, b2n, w3n, b3n,
                                                w2n16, neffs);
    // 4) edge MLPs -> m_vals, cinv   (LDS: 64KB weights + 4*8KB h1 tiles = 96KB)
    edge_mlp_kernel<<<256, 128, 98304, stream>>>(hs, neffs, coo, dis, mask, is_cent,
                                                 w1c, b1c, b2c, w3c, b3c,
                                                 w1k, b1k, b2k, w3k, b3k,
                                                 w2c16, m_vals, cinv);
    // 5) LDS-resident Euler integration (147456 B dynamic LDS, one WGP)
    euler_kernel<<<1, 1024, 4 * NSQ * sizeof(float), stream>>>(U0, m_vals, cinv,
                                                               steps, uRe, uIm);
    // 6) upsample to 2 x 768 x 768
    upsample_kernel<<<(768 * 768) / 256, 256, 0, stream>>>(uRe, uIm, Eys,
                                                           prof_re, prof_im,
                                                           (float*)d_out);
}
